// AdvancedGraphMatcher_82738249990887
// MI455X (gfx1250) — matrix operand order repaired and checked by reading, not compiled, not run on hardware
//
#include <hip/hip_runtime.h>
#include <hip/hip_bf16.h>
#include <math.h>

typedef __bf16 bf16_t;
typedef __attribute__((ext_vector_type(16))) __bf16 v16bf;
typedef __attribute__((ext_vector_type(8)))  __bf16 v8bf;
typedef __attribute__((ext_vector_type(8)))  float  v8f;

#define NN     512        // N1 == N2
#define EDGES  16384
#define NDIM   256
#define EDIM   64
#define HID    256
#define NHEADS 8
#define HDIM   32
#define LA_N   513

enum { FLAG_BIAS = 1, FLAG_RELU = 2, FLAG_ACCUM = 4 };

// ---------------------------------------------------------------------------
// WMMA helpers (wave32, 16x16x32 bf16 -> f32)
// A fragment: lane = M row (lane&15), lane>>4 selects K-halves per ISA layout.
// B is stored transposed [N][K] so each lane's fragment is two contiguous 16B chunks (K ks..ks+7, ks+16..ks+23).
// ---------------------------------------------------------------------------
__device__ __forceinline__ v8f wmma_bf16(v16bf a, v16bf b, v8f c) {
  return __builtin_amdgcn_wmma_f32_16x16x32_bf16(false, a, false, b,
                                                 (short)0, c, false, false);
}

__device__ __forceinline__ v16bf load_a_frag(const bf16_t* __restrict__ A,
                                             int lda, int mbase, int kb, int lane) {
  const int m  = mbase + (lane & 15);
  const int ks = kb + ((lane >> 4) << 3);      // 0 or 8
  const v8bf* p = reinterpret_cast<const v8bf*>(A + (size_t)m * lda + ks);
  v8bf lo = p[0];                               // K = ks .. ks+7
  v8bf hi = p[2];                               // K = ks+16 .. ks+23
  v16bf r;
#pragma unroll
  for (int j = 0; j < 8; ++j) { r[j] = lo[j]; r[j + 8] = hi[j]; }
  return r;
}

__device__ __forceinline__ v16bf load_b_frag(const bf16_t* __restrict__ Bt,
                                             int ldb, int nbase, int kb, int lane) {
  const int n  = nbase + (lane & 15);
  const int ks = kb + ((lane >> 4) << 3);      // 0 or 8
  const v8bf* p = reinterpret_cast<const v8bf*>(Bt + (size_t)n * ldb + ks);
  v8bf lo = p[0];
  v8bf hi = p[2];
  v16bf r;
#pragma unroll
  for (int j = 0; j < 8; ++j) { r[j] = lo[j]; r[j + 8] = hi[j]; }
  return r;
}

// ---------------------------------------------------------------------------
// Register-blocked GEMM: wave = 32x64 tile (2 A-frags x 4 B-frags, 8 WMMAs per
// K-step from 6 fragment loads -> ~2.7x the arithmetic intensity of 16x16).
// Requires M%32==0 && N%64==0 (true for all shapes here except attention P*V).
// ---------------------------------------------------------------------------
__global__ __launch_bounds__(256) void k_gemm_b(
    const bf16_t* __restrict__ A, const bf16_t* __restrict__ Bt,
    const float* __restrict__ bias, float* __restrict__ C,
    int M, int N, int K, int lda, int ldb, int ldc,
    long long sA, long long sB, long long sC, int flags)
{
  const int lane = threadIdx.x & 31;
  const int wave = threadIdx.x >> 5;
  const int tile = blockIdx.x * 8 + wave;
  const int tN   = N >> 6;
  if (tile >= (M >> 5) * tN) return;
  const int mt = tile / tN;
  const int nt = tile % tN;
  A  += (size_t)blockIdx.y * sA;
  Bt += (size_t)blockIdx.y * sB;
  C  += (size_t)blockIdx.y * sC;
  const int m0 = mt << 5;
  const int n0 = nt << 6;

  const v8f vz = {0.f, 0.f, 0.f, 0.f, 0.f, 0.f, 0.f, 0.f};
  v8f acc[2][4];
#pragma unroll
  for (int i = 0; i < 2; ++i)
#pragma unroll
    for (int j = 0; j < 4; ++j) acc[i][j] = vz;

  for (int kb = 0; kb < K; kb += 32) {
    v16bf a0 = load_a_frag(A, lda, m0,      kb, lane);
    v16bf a1 = load_a_frag(A, lda, m0 + 16, kb, lane);
#pragma unroll
    for (int j = 0; j < 4; ++j) {
      v16bf b = load_b_frag(Bt, ldb, n0 + (j << 4), kb, lane);
      acc[0][j] = wmma_bf16(a0, b, acc[0][j]);
      acc[1][j] = wmma_bf16(a1, b, acc[1][j]);
    }
  }
  const int nl = lane & 15;
  const int mo = (lane >> 4) << 3;
#pragma unroll
  for (int i = 0; i < 2; ++i) {
#pragma unroll
    for (int j = 0; j < 4; ++j) {
      const int n  = n0 + (j << 4) + nl;
      const float bv = (flags & FLAG_BIAS) ? bias[n] : 0.f;
      const int mrow = m0 + (i << 4) + mo;
#pragma unroll
      for (int v = 0; v < 8; ++v) {
        float x = acc[i][j][v] + bv;
        if (flags & FLAG_RELU) x = fmaxf(x, 0.f);
        float* cp = C + (size_t)(mrow + v) * ldc + n;
        if (flags & FLAG_ACCUM) x += *cp;
        *cp = x;
      }
    }
  }
}

// ---------------------------------------------------------------------------
// Fallback GEMM (wave = one 16x16 tile) for shapes with N%64!=0 (P*V, N=32).
// ---------------------------------------------------------------------------
__global__ __launch_bounds__(256) void k_gemm(
    const bf16_t* __restrict__ A, const bf16_t* __restrict__ Bt,
    const float* __restrict__ bias, float* __restrict__ C,
    int M, int N, int K, int lda, int ldb, int ldc,
    long long sA, long long sB, long long sC, int flags)
{
  const int lane   = threadIdx.x & 31;
  const int wave   = threadIdx.x >> 5;
  const int tile   = blockIdx.x * 8 + wave;
  const int tilesN = N >> 4;
  if (tile >= (M >> 4) * tilesN) return;
  const int mt = tile / tilesN;
  const int nt = tile % tilesN;
  A  += (size_t)blockIdx.y * sA;
  Bt += (size_t)blockIdx.y * sB;
  C  += (size_t)blockIdx.y * sC;

  v8f acc = {0.f, 0.f, 0.f, 0.f, 0.f, 0.f, 0.f, 0.f};
  for (int kb = 0; kb < K; kb += 32) {
    v16bf af = load_a_frag(A, lda, mt * 16, kb, lane);
    v16bf bf = load_b_frag(Bt, ldb, nt * 16, kb, lane);
    acc = wmma_bf16(af, bf, acc);
  }
  const int n  = nt * 16 + (lane & 15);
  const int m0 = mt * 16 + ((lane >> 4) << 3);
  const float bv = (flags & FLAG_BIAS) ? bias[n] : 0.f;
#pragma unroll
  for (int v = 0; v < 8; ++v) {
    float x = acc[v] + bv;
    if (flags & FLAG_RELU) x = fmaxf(x, 0.f);
    float* cp = C + (size_t)(m0 + v) * ldc + n;
    if (flags & FLAG_ACCUM) x += *cp;
    *cp = x;
  }
}

// ---------------------------------------------------------------------------
// Fused match head: per (i-tile, j) build h=relu(A[i,:]+Bb[j,:]) as bf16 A-frag
// in registers, contract with W2t (128x256) via WMMA, relu, dot w3, emit score.
// Never materializes the 512x512x256 tensor (268 MB avoided).
// ---------------------------------------------------------------------------
__global__ __launch_bounds__(256) void k_match_fused(
    const float* __restrict__ A, const float* __restrict__ Bb,
    const bf16_t* __restrict__ W2t, const float* __restrict__ w3,
    const float* __restrict__ b3, float* __restrict__ la, float invT)
{
  const int lane = threadIdx.x & 31;
  const int wave = threadIdx.x >> 5;
  const int gw   = blockIdx.x * 8 + wave;
  const int j  = gw >> 5;         // 0..511
  const int mt = gw & 31;         // i-tile
  if (j >= NN) return;

  const v8f vzero = {0.f, 0.f, 0.f, 0.f, 0.f, 0.f, 0.f, 0.f};
  v8f acc[8];
#pragma unroll
  for (int nf = 0; nf < 8; ++nf) acc[nf] = vzero;

  const int   mlane = mt * 16 + (lane & 15);
  const int   ksel  = (lane >> 4) << 3;
  const float* arow = A  + (size_t)mlane * NDIM;
  const float* brow = Bb + (size_t)j * NDIM;

  for (int kb = 0; kb < NDIM; kb += 32) {
    const float* ap = arow + kb + ksel;
    const float* bp = brow + kb + ksel;
    v16bf af;
#pragma unroll
    for (int t = 0; t < 8; ++t) {
      af[t]     = (bf16_t)fmaxf(ap[t]      + bp[t],      0.f);
      af[t + 8] = (bf16_t)fmaxf(ap[16 + t] + bp[16 + t], 0.f);
    }
#pragma unroll
    for (int nf = 0; nf < 8; ++nf) {
      v16bf bf = load_b_frag(W2t, NDIM, nf * 16, kb, lane);
      acc[nf] = wmma_bf16(af, bf, acc[nf]);
    }
  }
  // G = relu(acc); score[m] = sum_n G[m][n]*w3[n] + b3
  const int nlane = lane & 15;
  float part[8];
#pragma unroll
  for (int v = 0; v < 8; ++v) part[v] = 0.f;
#pragma unroll
  for (int nf = 0; nf < 8; ++nf) {
    const float w = w3[nf * 16 + nlane];
#pragma unroll
    for (int v = 0; v < 8; ++v) part[v] += fmaxf(acc[nf][v], 0.f) * w;
  }
#pragma unroll
  for (int off = 8; off >= 1; off >>= 1) {
#pragma unroll
    for (int v = 0; v < 8; ++v) part[v] += __shfl_xor(part[v], off, 32);
  }
  if (nlane == 0) {
    const int m0 = mt * 16 + ((lane >> 4) << 3);
    const float bias = b3[0];
#pragma unroll
    for (int v = 0; v < 8; ++v)
      la[(size_t)(m0 + v) * LA_N + j] = (part[v] + bias) * invT;
  }
}

// ---------------------------------------------------------------------------
// Elementwise / feature kernels
// ---------------------------------------------------------------------------
__global__ void k_cvt(const float* __restrict__ s, bf16_t* __restrict__ d, int n) {
  int i = blockIdx.x * blockDim.x + threadIdx.x;
  if (i < n) d[i] = (bf16_t)s[i];
}

__global__ void k_tcvt(const float* __restrict__ W, bf16_t* __restrict__ Wt,
                       int K, int N) {  // W[K][N] -> Wt[N][K] bf16
  int i = blockIdx.x * blockDim.x + threadIdx.x;
  if (i >= K * N) return;
  int n = i / K, k = i % K;
  Wt[i] = (bf16_t)W[(size_t)k * N + n];
}

__global__ void k_node_feats(const float* __restrict__ m, float* __restrict__ nf, int n) {
  int i = blockIdx.x * blockDim.x + threadIdx.x;
  if (i >= n) return;
  float x = m[i*5], y = m[i*5+1], th = m[i*5+2], q = m[i*5+3], t = m[i*5+4];
  nf[i*7+0] = x * (1.f/500.f);
  nf[i*7+1] = y * (1.f/500.f);
  nf[i*7+2] = cosf(th);
  nf[i*7+3] = sinf(th);
  nf[i*7+4] = th * 0.318309886f;   // th/pi
  nf[i*7+5] = q;
  nf[i*7+6] = t;
}

__global__ void k_edge_attr(const float* __restrict__ mi, const int* __restrict__ ei,
                            float* __restrict__ er, int E_) {
  int e = blockIdx.x * blockDim.x + threadIdx.x;
  if (e >= E_) return;
  int i = ei[e], j = ei[E_ + e];
  float dx = mi[j*5]   - mi[i*5];
  float dy = mi[j*5+1] - mi[i*5+1];
  float dist = sqrtf(dx*dx + dy*dy + 1e-12f);
  float ca = atan2f(dy, dx);
  float a1 = mi[i*5+2] - ca, a2 = mi[j*5+2] - ca;
  er[e*3+0] = dist;
  er[e*3+1] = atan2f(sinf(a1), cosf(a1));
  er[e*3+2] = atan2f(sinf(a2), cosf(a2));
}

// small-K dense layer (K=3 or 7), out = relu(X@W + b)
__global__ void k_lin_small(const float* __restrict__ X, const float* __restrict__ W,
                            const float* __restrict__ b, float* __restrict__ out,
                            int M, int K, int N) {
  int i = blockIdx.x * blockDim.x + threadIdx.x;
  if (i >= M * N) return;
  int mm = i / N, nn = i % N;
  float s = b[nn];
  for (int k = 0; k < K; ++k) s += X[(size_t)mm * K + k] * W[(size_t)k * N + nn];
  out[i] = fmaxf(s, 0.f);
}

// LayerNorm (wave per row), optional residual: out = LN(x + res)
__global__ void k_ln(const float* __restrict__ x, const float* __restrict__ res,
                     const float* __restrict__ g, const float* __restrict__ b,
                     float* __restrict__ out, int M, int D) {
  const int lane = threadIdx.x & 31;
  const int row  = blockIdx.x * 8 + (threadIdx.x >> 5);
  if (row >= M) return;
  const int cnt = D >> 5;            // D/32 (2 or 8)
  float vals[8];
  const float* xp = x + (size_t)row * D;
  const float* rp = res ? res + (size_t)row * D : nullptr;
  float s = 0.f;
  for (int t = 0; t < cnt; ++t) {
    float v = xp[lane + t * 32];
    if (rp) v += rp[lane + t * 32];
    vals[t] = v; s += v;
  }
#pragma unroll
  for (int o = 16; o >= 1; o >>= 1) s += __shfl_xor(s, o, 32);
  const float mean = s / (float)D;
  float vs = 0.f;
  for (int t = 0; t < cnt; ++t) { float d = vals[t] - mean; vs += d * d; }
#pragma unroll
  for (int o = 16; o >= 1; o >>= 1) vs += __shfl_xor(vs, o, 32);
  const float rstd = rsqrtf(vs / (float)D + 1e-5f);
  for (int t = 0; t < cnt; ++t) {
    int c = lane + t * 32;
    out[(size_t)row * D + c] = (vals[t] - mean) * rstd * g[c] + b[c];
  }
}

// msg_in = concat(x[ei[1]], x[ei[0]], ea) -> bf16 [E][576]
__global__ void k_msg_concat(const float* __restrict__ x, const float* __restrict__ ea,
                             const int* __restrict__ ei, bf16_t* __restrict__ Ab, int E_) {
  int i = blockIdx.x * blockDim.x + threadIdx.x;
  if (i >= E_ * 576) return;
  int e = i / 576, c = i % 576;
  float v;
  if (c < 256)       v = x[(size_t)ei[E_ + e] * NDIM + c];
  else if (c < 512)  v = x[(size_t)ei[e] * NDIM + (c - 256)];
  else               v = ea[(size_t)e * EDIM + (c - 512)];
  Ab[i] = (bf16_t)v;
}

__global__ void k_segsum(const float* __restrict__ m, const int* __restrict__ dst,
                         float* __restrict__ aggr, int E_) {
  int i = blockIdx.x * blockDim.x + threadIdx.x;
  if (i >= E_ * NDIM) return;
  int e = i >> 8, c = i & 255;
  atomicAdd(&aggr[(size_t)dst[e] * NDIM + c], m[i]);
}

// concat(x, aggr) -> bf16 [NN][512]
__global__ void k_ucat(const float* __restrict__ x, const float* __restrict__ aggr,
                       bf16_t* __restrict__ out, int M) {
  int i = blockIdx.x * blockDim.x + threadIdx.x;
  if (i >= M * 512) return;
  int r = i >> 9, c = i & 511;
  float v = (c < 256) ? x[(size_t)r * NDIM + c] : aggr[(size_t)r * NDIM + (c - 256)];
  out[i] = (bf16_t)v;
}

// split qkv into per-head bf16 Q[h][r][d], K[h][r][d], Vt[h][d][r]
__global__ void k_split_qkv(const float* __restrict__ qQ, const float* __restrict__ qKV,
                            bf16_t* __restrict__ Qb, bf16_t* __restrict__ Kb,
                            bf16_t* __restrict__ Vtb) {
  int i = blockIdx.x * blockDim.x + threadIdx.x;
  if (i >= NHEADS * NN * HDIM) return;
  int h = i / (NN * HDIM);
  int rem = i % (NN * HDIM);
  int r = rem / HDIM, d = rem % HDIM;
  Qb[i] = (bf16_t)qQ[(size_t)r * 768 + h * HDIM + d];
  Kb[i] = (bf16_t)qKV[(size_t)r * 768 + 256 + h * HDIM + d];
  Vtb[(size_t)h * HDIM * NN + (size_t)d * NN + r] =
      (bf16_t)qKV[(size_t)r * 768 + 512 + h * HDIM + d];
}

// softmax over rows of S (scale applied), emit bf16 P
__global__ void k_softmax(const float* __restrict__ S, bf16_t* __restrict__ P,
                          int R, int L, float scale) {
  const int lane = threadIdx.x & 31;
  const int row  = blockIdx.x * 8 + (threadIdx.x >> 5);
  if (row >= R) return;
  const float* p = S + (size_t)row * L;
  const int cnt = L >> 5;            // 16
  float vals[16];
  float mx = -3.0e38f;
  for (int t = 0; t < cnt; ++t) { vals[t] = p[lane + t * 32] * scale; mx = fmaxf(mx, vals[t]); }
#pragma unroll
  for (int o = 16; o >= 1; o >>= 1) mx = fmaxf(mx, __shfl_xor(mx, o, 32));
  float s = 0.f;
  for (int t = 0; t < cnt; ++t) { vals[t] = __expf(vals[t] - mx); s += vals[t]; }
#pragma unroll
  for (int o = 16; o >= 1; o >>= 1) s += __shfl_xor(s, o, 32);
  const float inv = 1.f / s;
  for (int t = 0; t < cnt; ++t)
    P[(size_t)row * L + lane + t * 32] = (bf16_t)(vals[t] * inv);
}

__global__ void k_la_init(float* __restrict__ la, const float* __restrict__ dustbin,
                          float invT) {
  int i = blockIdx.x * blockDim.x + threadIdx.x;
  if (i < LA_N * LA_N) la[i] = dustbin[0] * invT;
}

__global__ void k_lse_rows(float* __restrict__ la) {
  const int lane = threadIdx.x & 31;
  const int row  = blockIdx.x * 8 + (threadIdx.x >> 5);
  if (row >= LA_N) return;
  float* p = la + (size_t)row * LA_N;
  float mx = -3.0e38f;
  for (int e = lane; e < LA_N; e += 32) mx = fmaxf(mx, p[e]);
#pragma unroll
  for (int o = 16; o >= 1; o >>= 1) mx = fmaxf(mx, __shfl_xor(mx, o, 32));
  float s = 0.f;
  for (int e = lane; e < LA_N; e += 32) s += __expf(p[e] - mx);
#pragma unroll
  for (int o = 16; o >= 1; o >>= 1) s += __shfl_xor(s, o, 32);
  const float lse = mx + __logf(s);
  for (int e = lane; e < LA_N; e += 32) p[e] -= lse;
}

__global__ void k_lse_cols(float* __restrict__ la) {
  const int lane = threadIdx.x & 31;
  const int col  = blockIdx.x * 8 + (threadIdx.x >> 5);
  if (col >= LA_N) return;
  float* p = la + col;
  float mx = -3.0e38f;
  for (int e = lane; e < LA_N; e += 32) mx = fmaxf(mx, p[(size_t)e * LA_N]);
#pragma unroll
  for (int o = 16; o >= 1; o >>= 1) mx = fmaxf(mx, __shfl_xor(mx, o, 32));
  float s = 0.f;
  for (int e = lane; e < LA_N; e += 32) s += __expf(p[(size_t)e * LA_N] - mx);
#pragma unroll
  for (int o = 16; o >= 1; o >>= 1) s += __shfl_xor(s, o, 32);
  const float lse = mx + __logf(s);
  for (int e = lane; e < LA_N; e += 32) p[(size_t)e * LA_N] -= lse;
}

__global__ void k_exp(const float* __restrict__ la, float* __restrict__ out, int n) {
  int i = blockIdx.x * blockDim.x + threadIdx.x;
  if (i < n) out[i] = __expf(la[i]);
}

// ---------------------------------------------------------------------------
// Host orchestration
// ---------------------------------------------------------------------------
extern "C" void kernel_launch(void* const* d_in, const int* in_sizes, int n_in,
                              void* d_out, int out_size, void* d_ws, size_t ws_size,
                              hipStream_t stream) {
  (void)in_sizes; (void)n_in; (void)out_size; (void)ws_size;
  const float* minu1 = (const float*)d_in[0];
  const float* minu2 = (const float*)d_in[1];
  const int*   ei1   = (const int*)d_in[2];
  const int*   ei2   = (const int*)d_in[3];

  // params flattened in sorted pytree order after the 4 tensor inputs
  const float *cOutW[3], *cOutB[3], *cQkvW[3], *cQkvB[3];
  for (int c = 0; c < 3; ++c) {
    cOutW[c] = (const float*)d_in[4 + 4*c + 0];
    cOutB[c] = (const float*)d_in[4 + 4*c + 1];
    cQkvW[c] = (const float*)d_in[4 + 4*c + 2];
    cQkvB[c] = (const float*)d_in[4 + 4*c + 3];
  }
  const float* dustbin = (const float*)d_in[16];
  const float* eL1W = (const float*)d_in[17];
  const float* eL1b = (const float*)d_in[18];
  const float* eL2W = (const float*)d_in[19];
  const float* eL2b = (const float*)d_in[20];
  const float* eLnB = (const float*)d_in[21];
  const float* eLnG = (const float*)d_in[22];
  const float *gLnB[6], *gLnG[6], *gM1W[6], *gM1B[6], *gM2W[6], *gM2B[6],
              *gM3W[6], *gM3B[6], *gU1W[6], *gU1B[6], *gU2W[6], *gU2B[6];
  for (int l = 0; l < 6; ++l) {
    int b = 23 + 12 * l;
    gLnB[l] = (const float*)d_in[b+0];  gLnG[l] = (const float*)d_in[b+1];
    gM1W[l] = (const float*)d_in[b+2];  gM1B[l] = (const float*)d_in[b+3];
    gM2W[l] = (const float*)d_in[b+4];  gM2B[l] = (const float*)d_in[b+5];
    gM3W[l] = (const float*)d_in[b+6];  gM3B[l] = (const float*)d_in[b+7];
    gU1W[l] = (const float*)d_in[b+8];  gU1B[l] = (const float*)d_in[b+9];
    gU2W[l] = (const float*)d_in[b+10]; gU2B[l] = (const float*)d_in[b+11];
  }
  const float* mL1W = (const float*)d_in[95];
  const float* mL1b = (const float*)d_in[96];
  const float* mL2W = (const float*)d_in[97];
  const float* mL3W = (const float*)d_in[99];
  const float* mL3b = (const float*)d_in[100];
  const float* nL1W = (const float*)d_in[101];
  const float* nL1b = (const float*)d_in[102];
  const float* nL2W = (const float*)d_in[103];
  const float* nL2b = (const float*)d_in[104];
  const float* nLnB = (const float*)d_in[105];
  const float* nLnG = (const float*)d_in[106];
  const float *sOutW[6], *sOutB[6], *sQkvW[6], *sQkvB[6];
  for (int l = 0; l < 6; ++l) {
    int b = 107 + 4 * l;
    sOutW[l] = (const float*)d_in[b+0]; sOutB[l] = (const float*)d_in[b+1];
    sQkvW[l] = (const float*)d_in[b+2]; sQkvB[l] = (const float*)d_in[b+3];
  }

  // ---- workspace bump allocator ----
  char* cur = (char*)d_ws;
  auto alloc = [&](size_t bytes) -> void* {
    void* p = (void*)cur;
    cur += (bytes + 255) & ~(size_t)255;
    return p;
  };
  // bf16 weight buffers (transposed [N][K])
  bf16_t* nodeL2t = (bf16_t*)alloc((size_t)256*256*2);
  bf16_t* edgeL2t = (bf16_t*)alloc((size_t)64*256*2);
  bf16_t *gM1t[6], *gM2t[6], *gM3t[6], *gU1t[6], *gU2t[6];
  for (int l = 0; l < 6; ++l) {
    gM1t[l] = (bf16_t*)alloc((size_t)256*576*2);
    gM2t[l] = (bf16_t*)alloc((size_t)256*256*2);
    gM3t[l] = (bf16_t*)alloc((size_t)256*256*2);
    gU1t[l] = (bf16_t*)alloc((size_t)256*512*2);
    gU2t[l] = (bf16_t*)alloc((size_t)256*256*2);
  }
  bf16_t *sQkvT[6], *sOutT[6], *cQkvT[3], *cOutT[3];
  for (int l = 0; l < 6; ++l) {
    sQkvT[l] = (bf16_t*)alloc((size_t)768*256*2);
    sOutT[l] = (bf16_t*)alloc((size_t)256*256*2);
  }
  for (int c = 0; c < 3; ++c) {
    cQkvT[c] = (bf16_t*)alloc((size_t)768*256*2);
    cOutT[c] = (bf16_t*)alloc((size_t)256*256*2);
  }
  bf16_t* WaT = (bf16_t*)alloc((size_t)256*256*2);
  bf16_t* WbT = (bf16_t*)alloc((size_t)256*256*2);
  bf16_t* W2t = (bf16_t*)alloc((size_t)128*256*2);
  // activations
  float* nf1 = (float*)alloc((size_t)NN*7*4);
  float* nf2 = (float*)alloc((size_t)NN*7*4);
  float* er1 = (float*)alloc((size_t)EDGES*3*4);
  float* er2 = (float*)alloc((size_t)EDGES*3*4);
  float* ea1 = (float*)alloc((size_t)EDGES*EDIM*4);
  float* ea2 = (float*)alloc((size_t)EDGES*EDIM*4);
  float* f1  = (float*)alloc((size_t)NN*NDIM*4);
  float* f2  = (float*)alloc((size_t)NN*NDIM*4);
  float* hA  = (float*)alloc((size_t)EDGES*HID*4);
  float* hB  = (float*)alloc((size_t)EDGES*HID*4);
  bf16_t* bigbf = (bf16_t*)alloc((size_t)EDGES*576*2);
  float* aggr = (float*)alloc((size_t)NN*NDIM*4);
  bf16_t* cat512 = (bf16_t*)alloc((size_t)NN*512*2);
  bf16_t* xb  = (bf16_t*)alloc((size_t)NN*NDIM*2);
  float* Oc   = (float*)alloc((size_t)NN*NDIM*4);
  float* uBuf = (float*)alloc((size_t)NN*NDIM*4);
  float* t0   = (float*)alloc((size_t)NN*768*4);
  float* t1   = (float*)alloc((size_t)NN*768*4);
  bf16_t* Qb  = (bf16_t*)alloc((size_t)NHEADS*NN*HDIM*2);
  bf16_t* Kb  = (bf16_t*)alloc((size_t)NHEADS*NN*HDIM*2);
  bf16_t* Vtb = (bf16_t*)alloc((size_t)NHEADS*NN*HDIM*2);
  float* Sbuf = (float*)alloc((size_t)NHEADS*NN*NN*4);
  bf16_t* Pb  = (bf16_t*)alloc((size_t)NHEADS*NN*NN*2);
  float* mA   = (float*)alloc((size_t)NN*NDIM*4);
  float* mBv  = (float*)alloc((size_t)NN*NDIM*4);
  float* la   = (float*)alloc((size_t)LA_N*LA_N*4);

  // ---- launch helpers ----
  auto cvt = [&](const float* s, bf16_t* d, int n) {
    k_cvt<<<(n + 255) / 256, 256, 0, stream>>>(s, d, n);
  };
  auto tcvt = [&](const float* W, bf16_t* Wt, int K, int N) {
    k_tcvt<<<(K * N + 255) / 256, 256, 0, stream>>>(W, Wt, K, N);
  };
  auto gemm = [&](const bf16_t* A, const bf16_t* Bt, const float* bias, float* C,
                  int M, int N, int K, int lda, int ldb, int ldc,
                  long long sA, long long sB, long long sC, int batch, int flags) {
    if ((M & 31) == 0 && (N & 63) == 0) {
      int tiles = (M / 32) * (N / 64);
      dim3 grid((tiles + 7) / 8, batch);
      k_gemm_b<<<grid, 256, 0, stream>>>(A, Bt, bias, C, M, N, K, lda, ldb, ldc,
                                         sA, sB, sC, flags);
    } else {
      int tiles = (M / 16) * (N / 16);
      dim3 grid((tiles + 7) / 8, batch);
      k_gemm<<<grid, 256, 0, stream>>>(A, Bt, bias, C, M, N, K, lda, ldb, ldc,
                                       sA, sB, sC, flags);
    }
  };
  auto ln = [&](const float* x, const float* res, const float* g, const float* b,
                float* out, int M, int D) {
    k_ln<<<(M + 7) / 8, 256, 0, stream>>>(x, res, g, b, out, M, D);
  };
  auto attention = [&](const float* qQ, const float* qKV, const bf16_t* outT,
                       const float* outB, float* f) {
    k_split_qkv<<<(NHEADS * NN * HDIM + 255) / 256, 256, 0, stream>>>(qQ, qKV, Qb, Kb, Vtb);
    gemm(Qb, Kb, nullptr, Sbuf, NN, NN, HDIM, HDIM, HDIM, NN,
         (long long)NN * HDIM, (long long)NN * HDIM, (long long)NN * NN, NHEADS, 0);
    k_softmax<<<NHEADS * NN / 8, 256, 0, stream>>>(Sbuf, Pb, NHEADS * NN, NN, 0.1767766953f);
    gemm(Pb, Vtb, nullptr, Oc, NN, HDIM, NN, NN, NN, NDIM,
         (long long)NN * NN, (long long)HDIM * NN, (long long)HDIM, NHEADS, 0);
    cvt(Oc, xb, NN * NDIM);
    gemm(xb, outT, outB, f, NN, NDIM, NDIM, NDIM, NDIM, NDIM,
         0, 0, 0, 1, FLAG_BIAS | FLAG_ACCUM);
  };

  // ---- weight transpose/convert (once per launch) ----
  tcvt(nL2W, nodeL2t, 256, 256);
  tcvt(eL2W, edgeL2t, 256, 64);
  for (int l = 0; l < 6; ++l) {
    tcvt(gM1W[l], gM1t[l], 576, 256);
    tcvt(gM2W[l], gM2t[l], 256, 256);
    tcvt(gM3W[l], gM3t[l], 256, 256);
    tcvt(gU1W[l], gU1t[l], 512, 256);
    tcvt(gU2W[l], gU2t[l], 256, 256);
    tcvt(sQkvW[l], sQkvT[l], 256, 768);
    tcvt(sOutW[l], sOutT[l], 256, 256);
  }
  for (int c = 0; c < 3; ++c) {
    tcvt(cQkvW[c], cQkvT[c], 256, 768);
    tcvt(cOutW[c], cOutT[c], 256, 256);
  }
  tcvt(mL1W,             WaT, 256, 256);   // rows   0..255 of l1.W
  tcvt(mL1W + 256 * 256, WbT, 256, 256);   // rows 256..511 of l1.W
  tcvt(mL2W, W2t, 256, 128);

  // ---- encoders ----
  k_node_feats<<<(NN + 255) / 256, 256, 0, stream>>>(minu1, nf1, NN);
  k_node_feats<<<(NN + 255) / 256, 256, 0, stream>>>(minu2, nf2, NN);
  k_edge_attr<<<(EDGES + 255) / 256, 256, 0, stream>>>(minu1, ei1, er1, EDGES);
  k_edge_attr<<<(EDGES + 255) / 256, 256, 0, stream>>>(minu2, ei2, er2, EDGES);

  // edge encoder (both graphs)
  const float* ers[2] = {er1, er2};
  float* eas[2] = {ea1, ea2};
  for (int g = 0; g < 2; ++g) {
    k_lin_small<<<(EDGES * HID + 255) / 256, 256, 0, stream>>>(ers[g], eL1W, eL1b, hA,
                                                               EDGES, 3, HID);
    cvt(hA, bigbf, EDGES * HID);
    gemm(bigbf, edgeL2t, eL2b, eas[g], EDGES, EDIM, HID, HID, HID, EDIM,
         0, 0, 0, 1, FLAG_BIAS);
    ln(eas[g], nullptr, eLnG, eLnB, eas[g], EDGES, EDIM);
  }
  // node encoder (both graphs)
  const float* nfs[2] = {nf1, nf2};
  float* fs[2] = {f1, f2};
  for (int g = 0; g < 2; ++g) {
    k_lin_small<<<(NN * HID + 255) / 256, 256, 0, stream>>>(nfs[g], nL1W, nL1b, Oc,
                                                            NN, 7, HID);
    cvt(Oc, xb, NN * HID);
    gemm(xb, nodeL2t, nL2b, fs[g], NN, NDIM, HID, HID, HID, NDIM,
         0, 0, 0, 1, FLAG_BIAS);
    ln(fs[g], nullptr, nLnG, nLnB, fs[g], NN, NDIM);
  }

  // ---- GNN + attention stack ----
  auto gnn_conv = [&](float* x, const int* ei, const float* ea, int l) {
    k_msg_concat<<<(EDGES * 576 + 255) / 256, 256, 0, stream>>>(x, ea, ei, bigbf, EDGES);
    gemm(bigbf, gM1t[l], gM1B[l], hA, EDGES, HID, 576, 576, 576, HID,
         0, 0, 0, 1, FLAG_BIAS | FLAG_RELU);
    cvt(hA, bigbf, EDGES * HID);
    gemm(bigbf, gM2t[l], gM2B[l], hB, EDGES, HID, HID, HID, HID, HID,
         0, 0, 0, 1, FLAG_BIAS | FLAG_RELU);
    cvt(hB, bigbf, EDGES * HID);
    gemm(bigbf, gM3t[l], gM3B[l], hA, EDGES, NDIM, HID, HID, HID, NDIM,
         0, 0, 0, 1, FLAG_BIAS);
    hipMemsetAsync(aggr, 0, (size_t)NN * NDIM * sizeof(float), stream);
    k_segsum<<<(EDGES * NDIM + 255) / 256, 256, 0, stream>>>(hA, ei + EDGES, aggr, EDGES);
    k_ucat<<<(NN * 512 + 255) / 256, 256, 0, stream>>>(x, aggr, cat512, NN);
    gemm(cat512, gU1t[l], gU1B[l], Oc, NN, HID, 512, 512, 512, HID,
         0, 0, 0, 1, FLAG_BIAS | FLAG_RELU);
    cvt(Oc, xb, NN * HID);
    gemm(xb, gU2t[l], gU2B[l], uBuf, NN, NDIM, HID, HID, HID, NDIM,
         0, 0, 0, 1, FLAG_BIAS);
    ln(uBuf, x, gLnG[l], gLnB[l], x, NN, NDIM);
  };

  for (int l = 0; l < 6; ++l) {
    gnn_conv(f1, ei1, ea1, l);
    gnn_conv(f2, ei2, ea2, l);
    // self attention: f += mha(f, f)
    for (int g = 0; g < 2; ++g) {
      float* f = fs[g];
      cvt(f, xb, NN * NDIM);
      gemm(xb, sQkvT[l], sQkvB[l], t0, NN, 768, NDIM, NDIM, NDIM, 768,
           0, 0, 0, 1, FLAG_BIAS);
      attention(t0, t0, sOutT[l], sOutB[l], f);
    }
    if (l % 2 == 1) {
      int c = l / 2;
      cvt(f1, xb, NN * NDIM);
      gemm(xb, cQkvT[c], cQkvB[c], t0, NN, 768, NDIM, NDIM, NDIM, 768,
           0, 0, 0, 1, FLAG_BIAS);
      cvt(f2, xb, NN * NDIM);
      gemm(xb, cQkvT[c], cQkvB[c], t1, NN, 768, NDIM, NDIM, NDIM, 768,
           0, 0, 0, 1, FLAG_BIAS);
      attention(t0, t1, cOutT[c], cOutB[c], f1);   // f1 += mha(f1, f2)
      attention(t1, t0, cOutT[c], cOutB[c], f2);   // f2 += mha(f2, f1)
    }
  }

  // ---- match head (fused) + Sinkhorn ----
  cvt(f1, xb, NN * NDIM);
  gemm(xb, WaT, nullptr, mA, NN, NDIM, NDIM, NDIM, NDIM, NDIM, 0, 0, 0, 1, 0);
  cvt(f2, xb, NN * NDIM);
  gemm(xb, WbT, mL1b, mBv, NN, NDIM, NDIM, NDIM, NDIM, NDIM, 0, 0, 0, 1, FLAG_BIAS);

  const float invT = 10.0f;   // 1 / TEMP
  k_la_init<<<(LA_N * LA_N + 255) / 256, 256, 0, stream>>>(la, dustbin, invT);
  k_match_fused<<<2048, 256, 0, stream>>>(mA, mBv, W2t, mL3W, mL3b, la, invT);

  for (int it = 0; it < 20; ++it) {
    k_lse_rows<<<(LA_N + 7) / 8, 256, 0, stream>>>(la);
    k_lse_cols<<<(LA_N + 7) / 8, 256, 0, stream>>>(la);
  }
  k_exp<<<(LA_N * LA_N + 255) / 256, 256, 0, stream>>>(la, (float*)d_out, LA_N * LA_N);
}
